// MultiverseAttention_15814069584143
// MI455X (gfx1250) — matrix operand order repaired and checked by reading, not compiled, not run on hardware
//
#include <hip/hip_runtime.h>
#include <hip/hip_bf16.h>

// ---------------- problem constants ----------------
#define BATCH     2
#define SEQ       2048
#define HIDDEN    1024
#define NHEADS    16
#define HEADDIM   64
#define OVD       32
#define PVD       32
#define ROWS      (BATCH*SEQ)   // 4096
#define BH        (BATCH*NHEADS)
#define SCALE_F   0.125f

// d_out layout (fp32 elements): out | attn | phases
#define OFF_ATTN   (ROWS*HIDDEN)
#define OFF_PHASES (OFF_ATTN + BH*SEQ*SEQ)

// workspace layout (bytes)
#define XH_OFF    ((size_t)0)
#define WCAT_OFF  ((size_t)8388608)
#define WOT_OFF   ((size_t)14680064)
#define QH_OFF    ((size_t)16777216)
#define KH_OFF    ((size_t)25165824)
#define VT_OFF    ((size_t)33554432)   // (B,H,64,S) f16 transposed
#define CTX_OFF   ((size_t)41943040)

#define WPB 8

// fused-attention LDS layout (floats)
#define LDSTRIDE   2056                 // 2048 + 8 pad: rows stay 32B aligned
#define SC_FLOATS  (16*LDSTRIDE)        // 32896
#define RED_OFF    SC_FLOATS            // 16x16 reduction partials
#define RINV_OFF   (SC_FLOATS + 256)    // 16 row scalars
#define SMEM_FLOATS (SC_FLOATS + 256 + 16)
#define SMEM_BYTES  (SMEM_FLOATS * 4)   // 132,672 B

typedef __attribute__((ext_vector_type(16))) _Float16 v16h;
typedef __attribute__((ext_vector_type(8)))  _Float16 v8h;
typedef __attribute__((ext_vector_type(8)))  float    v8f;
typedef __attribute__((ext_vector_type(4)))  float    v4f;

union HFrag { v16h v; v8h h[2]; };

// A fragment 16x32 f16, row-major, base -> A[row0][k0]
__device__ inline v16h load_a_f16(const _Float16* __restrict__ base, int lda, int lane) {
    int row = lane & 15, hi = lane >> 4;
    const _Float16* p = base + (size_t)row * lda + hi * 8;
    HFrag f;
    f.h[0] = *(const v8h*)(p);
    f.h[1] = *(const v8h*)(p + 16);
    return f.v;
}

// B fragment 32x16 f16, N-major: BT[n][k], base -> BT[n0][k0]
__device__ inline v16h load_b_nt(const _Float16* __restrict__ base, int ldb, int lane) {
    int col = lane & 15, hi = lane >> 4;
    const _Float16* p = base + (size_t)col * ldb + hi * 8;
    HFrag f;
    f.h[0] = *(const v8h*)(p);
    f.h[1] = *(const v8h*)(p + 16);
    return f.v;
}

// A fragment from fp32 LDS, converted to f16. base -> A[row0][k0], ldf floats
__device__ inline v16h load_a_lds_f32(const float* base, int ldf, int lane) {
    int row = lane & 15, hi = lane >> 4;
    const float* p = base + row * ldf + hi * 8;
    v4f x0 = *(const v4f*)(p);
    v4f x1 = *(const v4f*)(p + 4);
    v4f x2 = *(const v4f*)(p + 16);
    v4f x3 = *(const v4f*)(p + 20);
    HFrag f;
#pragma unroll
    for (int i = 0; i < 4; ++i) {
        f.h[0][i]     = (_Float16)x0[i];
        f.h[0][4 + i] = (_Float16)x1[i];
        f.h[1][i]     = (_Float16)x2[i];
        f.h[1][4 + i] = (_Float16)x3[i];
    }
    return f.v;
}

__device__ inline v8f wmma16x16x32(v16h a, v16h b, v8f c) {
    return __builtin_amdgcn_wmma_f32_16x16x32_f16(false, a, false, b, (short)0, c, false, false);
}

// ---------------- packing kernels ----------------
__global__ void k_convert_x(const float* __restrict__ x, _Float16* __restrict__ xh) {
    int gid = blockIdx.x * 256 + threadIdx.x;
    xh[gid] = (_Float16)x[gid];
}

__global__ void k_pack_wcat(const float* __restrict__ Wq, const float* __restrict__ Wk,
                            const float* __restrict__ Wv, const float* __restrict__ Wp,
                            _Float16* __restrict__ wcat) {
    int gid = blockIdx.x * 256 + threadIdx.x;   // 16*192*1024
    int d = gid & 1023;
    int n = (gid >> 10) % 192;
    int h = gid / (192 * 1024);
    int seg = n >> 6;
    int c   = n & 63;
    float v;
    if (c < 32) {
        const float* W = (seg == 0) ? Wq : (seg == 1) ? Wk : Wv;
        v = W[(size_t)d * (OVD * NHEADS) + h * OVD + c];
    } else {
        v = Wp[((size_t)h * HIDDEN + d) * (3 * PVD) + seg * PVD + (c - 32)];
    }
    wcat[gid] = (_Float16)v;
}

__global__ void k_pack_wot(const float* __restrict__ Wo, _Float16* __restrict__ wot) {
    int gid = blockIdx.x * 256 + threadIdx.x;
    int n = gid >> 10, d = gid & 1023;
    wot[gid] = (_Float16)Wo[(size_t)d * HIDDEN + n];
}

__global__ void k_phases(const float* __restrict__ x, const float* __restrict__ Wph,
                         const float* __restrict__ bph, float* __restrict__ ph) {
    int gid = blockIdx.x * 256 + threadIdx.x;
    int r = gid >> 4, h = gid & 15;
    const float* xr = x + (size_t)r * HIDDEN;
    float s = 0.f;
#pragma unroll 4
    for (int d = 0; d < HIDDEN; ++d) s += xr[d] * Wph[d * NHEADS + h];
    ph[gid] = s + bph[h];
}

// ---------------- fused QKV + private projection GEMM ----------------
__global__ void __launch_bounds__(256, 2)
k_qkv_gemm(const _Float16* __restrict__ xh, const _Float16* __restrict__ wcat,
           const float* __restrict__ bq, const float* __restrict__ bk,
           const float* __restrict__ bv, const float* __restrict__ bp,
           _Float16* __restrict__ qh, _Float16* __restrict__ kh,
           _Float16* __restrict__ vt) {
    int lane = threadIdx.x & 31;
    int wid  = blockIdx.x * WPB + (threadIdx.x >> 5);
    int seg = wid % 3;
    int h   = (wid / 3) & 15;
    int rt  = wid / 48;
    int r0  = rt * 16;
    const _Float16* Abase = xh + (size_t)r0 * HIDDEN;
    const _Float16* Bbase = wcat + ((size_t)h * 192 + seg * 64) * HIDDEN;
    v8f c0 = {}, c1 = {}, c2 = {}, c3 = {};
    v16h a  = load_a_f16(Abase, HIDDEN, lane);
    v16h b0 = load_b_nt(Bbase,                        HIDDEN, lane);
    v16h b1 = load_b_nt(Bbase + (size_t)16 * HIDDEN,  HIDDEN, lane);
    v16h b2 = load_b_nt(Bbase + (size_t)32 * HIDDEN,  HIDDEN, lane);
    v16h b3 = load_b_nt(Bbase + (size_t)48 * HIDDEN,  HIDDEN, lane);
    for (int k0 = 32; k0 < HIDDEN; k0 += 32) {
        v16h an = load_a_f16(Abase + k0, HIDDEN, lane);
        v16h n0 = load_b_nt(Bbase + k0,                       HIDDEN, lane);
        c0 = wmma16x16x32(a, b0, c0);
        v16h n1 = load_b_nt(Bbase + (size_t)16 * HIDDEN + k0, HIDDEN, lane);
        c1 = wmma16x16x32(a, b1, c1);
        v16h n2 = load_b_nt(Bbase + (size_t)32 * HIDDEN + k0, HIDDEN, lane);
        c2 = wmma16x16x32(a, b2, c2);
        v16h n3 = load_b_nt(Bbase + (size_t)48 * HIDDEN + k0, HIDDEN, lane);
        c3 = wmma16x16x32(a, b3, c3);
        a = an; b0 = n0; b1 = n1; b2 = n2; b3 = n3;
    }
    c0 = wmma16x16x32(a, b0, c0);
    c1 = wmma16x16x32(a, b1, c1);
    c2 = wmma16x16x32(a, b2, c2);
    c3 = wmma16x16x32(a, b3, c3);

    v8f cc[4] = {c0, c1, c2, c3};
    int hi = lane >> 4, nl = lane & 15;
#pragma unroll
    for (int j = 0; j < 4; ++j) {
#pragma unroll
        for (int i = 0; i < 8; ++i) {
            int r   = r0 + i + 8 * hi;
            int col = j * 16 + nl;
            float bias = (col < 32)
                ? ((seg == 0) ? bq : (seg == 1) ? bk : bv)[h * OVD + col]
                : bp[h * (3 * PVD) + seg * PVD + (col - 32)];
            float val = cc[j][i] + bias;
            int b = r >> 11, s = r & 2047;
            int bh = b * NHEADS + h;
            if (seg == 2)
                vt[((size_t)bh * HEADDIM + col) * SEQ + s] = (_Float16)val;
            else {
                _Float16* dst = (seg == 0) ? qh : kh;
                dst[((size_t)bh * SEQ + s) * HEADDIM + col] = (_Float16)val;
            }
        }
    }
}

// ---------------- fused attention: scores -> softmax -> P@V, per 16-row block ----------------
// grid: BH * 128 blocks of 256 threads; dynamic LDS = SMEM_BYTES
__global__ void __launch_bounds__(256, 2)
k_attn_fused(const _Float16* __restrict__ qh, const _Float16* __restrict__ kh,
             const _Float16* __restrict__ vt, const int* __restrict__ mask,
             const float* __restrict__ ph, float* __restrict__ attn,
             _Float16* __restrict__ ctx) {
    extern __shared__ float smem[];
    float* sc   = smem;               // [16][LDSTRIDE] scores / exp values
    float* red  = smem + RED_OFF;     // [16][16] reduction partials
    float* rinv = smem + RINV_OFF;    // [16] row max, then 1/sum

    int t    = threadIdx.x;
    int lane = t & 31;
    int w    = t >> 5;                // wave 0..7
    int rt   = blockIdx.x & 127;
    int bh   = blockIdx.x >> 7;
    int bb   = bh >> 4;
    int hh   = bh & 15;
    int r0   = rt * 16;
    int hi = lane >> 4, nl = lane & 15;

    // ---- phase 1: scores (16 x 2048) into LDS, wave w owns keys [w*256, w*256+256)
    const _Float16* Abase = qh + ((size_t)bh * SEQ + r0) * HEADDIM;
    v16h a0 = load_a_f16(Abase,      HEADDIM, lane);
    v16h a1 = load_a_f16(Abase + 32, HEADDIM, lane);
    const _Float16* Kb = kh + ((size_t)bh * SEQ + w * 256) * HEADDIM;
    v16h b0 = load_b_nt(Kb,      HEADDIM, lane);
    v16h b1 = load_b_nt(Kb + 32, HEADDIM, lane);
    for (int kt = 0; kt < 16; ++kt) {
        v16h cb0 = b0, cb1 = b1;
        if (kt < 15) {
            const _Float16* nKb = Kb + (size_t)(kt + 1) * 16 * HEADDIM;
            b0 = load_b_nt(nKb,      HEADDIM, lane);
            b1 = load_b_nt(nKb + 32, HEADDIM, lane);
        }
        v8f c = {};
        c = wmma16x16x32(a0, cb0, c);
        c = wmma16x16x32(a1, cb1, c);
        int col = w * 256 + kt * 16 + nl;
        int km  = mask[bb * SEQ + col];
        float* dst = sc + col;
#pragma unroll
        for (int i = 0; i < 8; ++i) {
            float val = c[i] * SCALE_F;
            if (km == 0) val = -__builtin_inff();
            dst[(i + 8 * hi) * LDSTRIDE] = val;
        }
    }
    __syncthreads();

    // ---- phase 2: row softmax in LDS; write normalized fp32 to d_out (only HBM touch)
    {
        int row  = t >> 4;
        int idx0 = t & 15;
        float* rp = sc + row * LDSTRIDE;
        float m = -__builtin_inff();
#pragma unroll 4
        for (int j = 0; j < 32; ++j) {
            v4f v = *(const v4f*)(rp + 4 * (idx0 + j * 16));
            m = fmaxf(m, fmaxf(fmaxf(v[0], v[1]), fmaxf(v[2], v[3])));
        }
        red[row * 16 + idx0] = m;
        __syncthreads();
        if (idx0 == 0) {
            float mm = red[row * 16];
#pragma unroll
            for (int k = 1; k < 16; ++k) mm = fmaxf(mm, red[row * 16 + k]);
            rinv[row] = mm;
        }
        __syncthreads();
        float mx = rinv[row];
        float s = 0.f;
#pragma unroll 4
        for (int j = 0; j < 32; ++j) {
            float* pp = rp + 4 * (idx0 + j * 16);
            v4f v = *(const v4f*)(pp);
            v4f e;
#pragma unroll
            for (int q = 0; q < 4; ++q) { e[q] = __expf(v[q] - mx); s += e[q]; }
            *(v4f*)(pp) = e;
        }
        red[row * 16 + idx0] = s;
        __syncthreads();
        if (idx0 == 0) {
            float ss = red[row * 16];
#pragma unroll
            for (int k = 1; k < 16; ++k) ss += red[row * 16 + k];
            rinv[row] = 1.0f / ss;
        }
        __syncthreads();
        float inv = rinv[row];
        float* gp = attn + ((size_t)bh * SEQ + r0 + row) * SEQ;
#pragma unroll 4
        for (int j = 0; j < 32; ++j) {
            int f4 = idx0 + j * 16;
            v4f e = *(const v4f*)(rp + 4 * f4);
            v4f o;
#pragma unroll
            for (int q = 0; q < 4; ++q) o[q] = e[q] * inv;
            *(v4f*)(gp + 4 * f4) = o;
        }
    }

    // ---- phase 3: out = P @ V (P = exp values in LDS, normalize in epilogue), split-K
    int tile  = w & 3;    // 16-dim column tile
    int khalf = w >> 2;   // key half: 0 -> [0,1024), 1 -> [1024,2048)
    v8f acc = {};
    const _Float16* Vb = vt + (size_t)bh * HEADDIM * SEQ + (size_t)(tile * 16) * SEQ + khalf * 1024;
    const float* Pb = sc + khalf * 1024;
    for (int k0 = 0; k0 < 1024; k0 += 32) {
        v16h a = load_a_lds_f32(Pb + k0, LDSTRIDE, lane);
        v16h b = load_b_nt(Vb + k0, SEQ, lane);
        acc = wmma16x16x32(a, b, acc);
    }
    __syncthreads();                     // all reads of sc done
    if (khalf == 1) {                    // park partials in (now free) score LDS
        float* ex = sc + (size_t)(tile * 32 + lane) * 8;
        *(v4f*)(ex)     = v4f{acc[0], acc[1], acc[2], acc[3]};
        *(v4f*)(ex + 4) = v4f{acc[4], acc[5], acc[6], acc[7]};
    }
    __syncthreads();
    if (khalf == 0) {
        const float* ex = sc + (size_t)(tile * 32 + lane) * 8;
        v4f p0 = *(const v4f*)(ex);
        v4f p1 = *(const v4f*)(ex + 4);
#pragma unroll
        for (int i = 0; i < 4; ++i) { acc[i] += p0[i]; acc[4 + i] += p1[i]; }
#pragma unroll
        for (int i = 0; i < 8; ++i) {
            int row = i + 8 * hi;
            int q   = r0 + row;
            int d   = tile * 16 + nl;
            float fct = rinv[row] * (1.0f + 0.1f * cosf(ph[(size_t)(bb * SEQ + q) * NHEADS + hh]));
            ctx[(size_t)(bb * SEQ + q) * HIDDEN + hh * HEADDIM + d] = (_Float16)(acc[i] * fct);
        }
    }
}

// ---------------- final projection: ctx @ Wo + bo ----------------
__global__ void __launch_bounds__(256, 2)
k_out_gemm(const _Float16* __restrict__ ctx, const _Float16* __restrict__ wot,
           const float* __restrict__ bo, float* __restrict__ out) {
    int lane = threadIdx.x & 31;
    int wid  = blockIdx.x * WPB + (threadIdx.x >> 5);
    int ct = wid & 15;
    int rt = wid >> 4;
    const _Float16* Abase = ctx + (size_t)(rt * 16) * HIDDEN;
    const _Float16* Bbase = wot + (size_t)(ct * 64) * HIDDEN;
    v8f c0 = {}, c1 = {}, c2 = {}, c3 = {};
    v16h a  = load_a_f16(Abase, HIDDEN, lane);
    v16h b0 = load_b_nt(Bbase,                       HIDDEN, lane);
    v16h b1 = load_b_nt(Bbase + (size_t)16 * HIDDEN, HIDDEN, lane);
    v16h b2 = load_b_nt(Bbase + (size_t)32 * HIDDEN, HIDDEN, lane);
    v16h b3 = load_b_nt(Bbase + (size_t)48 * HIDDEN, HIDDEN, lane);
    for (int k0 = 32; k0 < HIDDEN; k0 += 32) {
        v16h an = load_a_f16(Abase + k0, HIDDEN, lane);
        v16h n0 = load_b_nt(Bbase + k0,                       HIDDEN, lane);
        c0 = wmma16x16x32(a, b0, c0);
        v16h n1 = load_b_nt(Bbase + (size_t)16 * HIDDEN + k0, HIDDEN, lane);
        c1 = wmma16x16x32(a, b1, c1);
        v16h n2 = load_b_nt(Bbase + (size_t)32 * HIDDEN + k0, HIDDEN, lane);
        c2 = wmma16x16x32(a, b2, c2);
        v16h n3 = load_b_nt(Bbase + (size_t)48 * HIDDEN + k0, HIDDEN, lane);
        c3 = wmma16x16x32(a, b3, c3);
        a = an; b0 = n0; b1 = n1; b2 = n2; b3 = n3;
    }
    c0 = wmma16x16x32(a, b0, c0);
    c1 = wmma16x16x32(a, b1, c1);
    c2 = wmma16x16x32(a, b2, c2);
    c3 = wmma16x16x32(a, b3, c3);

    v8f cc[4] = {c0, c1, c2, c3};
    int hi = lane >> 4, nl = lane & 15;
#pragma unroll
    for (int j = 0; j < 4; ++j) {
#pragma unroll
        for (int i = 0; i < 8; ++i) {
            int r = rt * 16 + i + 8 * hi;
            int n = ct * 64 + j * 16 + nl;
            out[(size_t)r * HIDDEN + n] = cc[j][i] + bo[n];
        }
    }
}

extern "C" void kernel_launch(void* const* d_in, const int* in_sizes, int n_in,
                              void* d_out, int out_size, void* d_ws, size_t ws_size,
                              hipStream_t stream) {
    const float* x    = (const float*)d_in[0];
    const int*   mask = (const int*)  d_in[1];
    const float* Wq   = (const float*)d_in[2];
    const float* bq   = (const float*)d_in[3];
    const float* Wk   = (const float*)d_in[4];
    const float* bk   = (const float*)d_in[5];
    const float* Wv   = (const float*)d_in[6];
    const float* bv   = (const float*)d_in[7];
    const float* Wp   = (const float*)d_in[8];
    const float* bp   = (const float*)d_in[9];
    const float* Wph  = (const float*)d_in[10];
    const float* bph  = (const float*)d_in[11];
    const float* Wo   = (const float*)d_in[12];
    const float* bo   = (const float*)d_in[13];

    float* out_f  = (float*)d_out;
    float* attn_f = out_f + OFF_ATTN;
    float* ph_f   = out_f + OFF_PHASES;

    char* ws = (char*)d_ws;
    _Float16* xh   = (_Float16*)(ws + XH_OFF);
    _Float16* wcat = (_Float16*)(ws + WCAT_OFF);
    _Float16* wot  = (_Float16*)(ws + WOT_OFF);
    _Float16* qh   = (_Float16*)(ws + QH_OFF);
    _Float16* kh   = (_Float16*)(ws + KH_OFF);
    _Float16* vt   = (_Float16*)(ws + VT_OFF);
    _Float16* ctx  = (_Float16*)(ws + CTX_OFF);

    k_convert_x<<<(ROWS * HIDDEN) / 256, 256, 0, stream>>>(x, xh);
    k_pack_wcat<<<(NHEADS * 192 * HIDDEN) / 256, 256, 0, stream>>>(Wq, Wk, Wv, Wp, wcat);
    k_pack_wot<<<(HIDDEN * HIDDEN) / 256, 256, 0, stream>>>(Wo, wot);
    k_phases<<<(ROWS * NHEADS) / 256, 256, 0, stream>>>(x, Wph, bph, ph_f);

    k_qkv_gemm<<<(256 * NHEADS * 3) / WPB, 256, 0, stream>>>(xh, wcat, bq, bk, bv, bp, qh, kh, vt);
    k_attn_fused<<<BH * 128, 256, SMEM_BYTES, stream>>>(qh, kh, vt, mask, ph_f, attn_f, ctx);
    k_out_gemm<<<(256 * 16) / WPB, 256, 0, stream>>>(ctx, wot, bo, out_f);
}
